// TransformerLayer_22832046146216
// MI455X (gfx1250) — matrix operand order repaired and checked by reading, not compiled, not run on hardware
//
#include <hip/hip_runtime.h>

typedef __attribute__((ext_vector_type(16))) _Float16 v16h;
typedef __attribute__((ext_vector_type(8)))  _Float16 v8h;
typedef __attribute__((ext_vector_type(8)))  float    v8f;
typedef __attribute__((ext_vector_type(4)))  float    v4f;
typedef __attribute__((ext_vector_type(2)))  float    v2f;

#define DEV static __device__ __forceinline__

constexpr int B  = 8;
constexpr int S  = 1024;
constexpr int D  = 1024;
constexpr int H  = 16;
constexpr int DH = 64;
constexpr float LN_EPS = 1e-5f;
constexpr float SLOPE  = 0.01f;   // LeakyReLU negative_slope

// ---------------------------------------------------------------------------
// WMMA fragment helpers (contiguous k order k = 16*(lane/16) + s for both A and B, wave32)
// A 16x32 f16: lane m = lane&15; lanes<16 hold K = 0..15, lanes>=16
// hold K = 16..31  -> two contiguous 16B chunks at +base and +base+8.
DEV v16h load_afrag(const _Float16* rowk0, int lane) {
  const _Float16* p = rowk0 + ((lane & 16) ? 16 : 0);
  v8h lo = *(const v8h*)(p);
  v8h hi = *(const v8h*)(p + 8);
  v16h r;
#pragma unroll
  for (int i = 0; i < 8; ++i) { r[i] = lo[i]; r[i + 8] = hi[i]; }
  return r;
}
// B 32x16 f16: lane n = lane&15; lanes<16 hold K=0..15, lanes>=16 K=16..31
// -> one contiguous 32B chunk (column-major storage: col*ldK + k).
DEV v16h load_bfrag(const _Float16* colk0, int lane) {
  return *(const v16h*)(colk0 + ((lane & 16) ? 16 : 0));
}

DEV v8f wmma_f16(v16h a, v16h b, v8f c) {
  return __builtin_amdgcn_wmma_f32_16x16x32_f16(
      /*neg_a=*/false, a, /*neg_b=*/false, b,
      /*c_mod=*/(short)0, c, /*reuse_a=*/false, /*reuse_b=*/false);
}

DEV float redsum32(float v) {
#pragma unroll
  for (int m = 16; m >= 1; m >>= 1) v += __shfl_xor(v, m, 32);
  return v;
}

// ---- CDNA5 async global->LDS DMA (ASYNCcnt path, CDNA5 ISA) --------
// Moves 16B from the lane's global address to the lane's LDS byte address.
DEV void async_load_lds_b128(unsigned lds_byte_addr, const void* gaddr) {
  asm volatile("global_load_async_to_lds_b128 %0, %1, off"
               :: "v"(lds_byte_addr), "v"(gaddr)
               : "memory");
}
DEV void wait_async0() {
  asm volatile("s_wait_asynccnt 0" ::: "memory");
}
// Generic (flat) shared pointer -> LDS byte address: LDS aperture maps
// addr[31:0] directly to the LDS offset (ISA §10.2 aperture table).
DEV unsigned lds_addr_of(const void* p) {
  return (unsigned)(unsigned long long)p;
}

// ---------------------------------------------------------------------------
// 1) per-(b,h,s) row: qn = qh / max(||qh||,1e-8) as f16 (B,H,S,DH) and
//    vt = qh^T as f16 (B,H,DH,S). One wave per row, 2 elems/lane.
__global__ void __launch_bounds__(256)
prep_kernel(const float* __restrict__ q, _Float16* __restrict__ qn,
            _Float16* __restrict__ vt) {
  int w = threadIdx.x >> 5, lane = threadIdx.x & 31;
  int row = blockIdx.x * 8 + w;                 // flat (b,h,s)
  int b = row >> 14, h = (row >> 10) & (H - 1), s = row & (S - 1);
  const float* src = q + (size_t)(b * S + s) * D + h * DH;
  v2f v = *(const v2f*)(src + 2 * lane);
  float ss  = redsum32(v[0] * v[0] + v[1] * v[1]);
  float inv = 1.0f / fmaxf(sqrtf(ss), 1e-8f);
  size_t bh = (size_t)(b * H + h);
  _Float16* qr = qn + (bh * S + s) * DH;
  qr[2 * lane]     = (_Float16)(v[0] * inv);
  qr[2 * lane + 1] = (_Float16)(v[1] * inv);
  _Float16* vr = vt + bh * DH * S + s;
  vr[(size_t)(2 * lane) * S]     = (_Float16)v[0];
  vr[(size_t)(2 * lane + 1) * S] = (_Float16)v[1];
}

// 2) f32 weight -> f16, transposed to column-major (n*D + k) for B-fragments.
__global__ void __launch_bounds__(256)
conv_wt_kernel(const float* __restrict__ w, _Float16* __restrict__ wt) {
  int idx = blockIdx.x * 256 + threadIdx.x;
  int k = idx >> 10, n = idx & (D - 1);
  wt[(size_t)n * D + k] = (_Float16)w[idx];
}

// ---------------------------------------------------------------------------
// 3) Cosine-sim attention. One wave per (b,h,16-query block).
//    Scores are cosine sims in [-1,1], so softmax needs NO running max:
//    p = exp(s-1) can't over/underflow over S=1024 keys. The denominator is
//    computed with an extra WMMA (P @ ones) so it lands directly in C-layout
//    -- zero cross-lane shuffles in the whole inner loop.
//    O += P@V, Csum += 1@V (post-softmax masked_fill path), L += P@1.
__global__ void __launch_bounds__(32)
attn_kernel(const _Float16* __restrict__ qn, const _Float16* __restrict__ vt,
            const int* __restrict__ masks, float* __restrict__ attn) {
  __shared__ __align__(32) _Float16 Pl[16 * 32];   // P tile, C->A relayout
  int lane = threadIdx.x;
  int qb = blockIdx.x & 63;
  int h  = (blockIdx.x >> 6) & 15;
  int b  = blockIdx.x >> 10;
  size_t bh = (size_t)(b * H + h);
  const _Float16* qn_bh = qn + bh * (size_t)S * DH;
  const _Float16* vt_bh = vt + bh * (size_t)DH * S;
  int lm  = lane & 15;
  int hi8 = (lane & 16) ? 8 : 0;

  // Q fragments for the 16 query rows (two K=32 steps over DH=64)
  v16h aQ0 = load_afrag(qn_bh + (size_t)(qb * 16 + lm) * DH + 0,  lane);
  v16h aQ1 = load_afrag(qn_bh + (size_t)(qb * 16 + lm) * DH + 32, lane);
  v16h ones;                       // all-ones: valid as both A and B fragment
#pragma unroll
  for (int i = 0; i < 16; ++i) ones[i] = (_Float16)1.0f;

  v8f o0 = {}, o1 = {}, o2 = {}, o3 = {};
  v8f c0 = {}, c1 = {}, c2 = {}, c3 = {};
  v8f lsum = {};                   // row-sums of P, C-layout

  for (int kb = 0; kb < S; kb += 32) {
    // ---- scores: two 16-key halves, each accumulated over K=64 features
    v8f s0 = {}, s1 = {};
    {
      v16h bK;
      bK = load_bfrag(qn_bh + (size_t)(kb + lm) * DH + 0,  lane);
      s0 = wmma_f16(aQ0, bK, s0);
      bK = load_bfrag(qn_bh + (size_t)(kb + lm) * DH + 32, lane);
      s0 = wmma_f16(aQ1, bK, s0);
      bK = load_bfrag(qn_bh + (size_t)(kb + 16 + lm) * DH + 0,  lane);
      s1 = wmma_f16(aQ0, bK, s1);
      bK = load_bfrag(qn_bh + (size_t)(kb + 16 + lm) * DH + 32, lane);
      s1 = wmma_f16(aQ1, bK, s1);
    }
    // ---- p = exp(s - 1): fixed shift, no reductions needed
#pragma unroll
    for (int r = 0; r < 8; ++r) {
      float p0 = __expf(s0[r] - 1.0f);
      float p1 = __expf(s1[r] - 1.0f);
      int m = r + hi8;
      Pl[m * 32 + lm]      = (_Float16)p0;
      Pl[m * 32 + 16 + lm] = (_Float16)p1;
    }
    asm volatile("s_wait_dscnt 0" ::: "memory");  // LDS RAW fence (1 wave)
    v16h aP = load_afrag((const _Float16*)(Pl + lm * 32), lane);
    const _Float16* vb = vt_bh + (size_t)lm * S + kb;
    v16h vb0 = load_bfrag(vb + (size_t)(0 * 16) * S, lane);
    v16h vb1 = load_bfrag(vb + (size_t)(1 * 16) * S, lane);
    v16h vb2 = load_bfrag(vb + (size_t)(2 * 16) * S, lane);
    v16h vb3 = load_bfrag(vb + (size_t)(3 * 16) * S, lane);
    o0 = wmma_f16(aP, vb0, o0);   c0 = wmma_f16(ones, vb0, c0);
    o1 = wmma_f16(aP, vb1, o1);   c1 = wmma_f16(ones, vb1, c1);
    o2 = wmma_f16(aP, vb2, o2);   c2 = wmma_f16(ones, vb2, c2);
    o3 = wmma_f16(aP, vb3, o3);   c3 = wmma_f16(ones, vb3, c3);
    lsum = wmma_f16(aP, ones, lsum);   // denominator, lands in C-layout
  }

  // ---- writeback: softmax-normalized, or -1e9 * colsum(V) for masked rows
#pragma unroll
  for (int r = 0; r < 8; ++r) {
    int m = r + hi8;
    int row = qb * 16 + m;
    int msk = masks[b * S + row];
    float invl = 1.0f / lsum[r];
    float* dst = attn + (size_t)(b * S + row) * D + h * DH;
    dst[0 * 16 + lm] = msk ? o0[r] * invl : -1e9f * c0[r];
    dst[1 * 16 + lm] = msk ? o1[r] * invl : -1e9f * c1[r];
    dst[2 * 16 + lm] = msk ? o2[r] * invl : -1e9f * c2[r];
    dst[3 * 16 + lm] = msk ? o3[r] * invl : -1e9f * c3[r];
  }
}

// ---------------------------------------------------------------------------
// 4) C[M,1024] = A_f16[M,1024] @ W (given as W^T col-major f16) + bias.
//    Workgroup = 8 waves = 256x64 output tile; each wave owns 32x64 so every
//    B-fragment feeds two A-tiles (8 WMMAs per fragment set). The shared
//    64-col x 64-K weight tile (8KB) is staged into LDS with async
//    global->LDS DMA, double-buffered: DMA of tile i+1 overlaps the 16 WMMAs
//    of tile i; one wait+barrier per 16 WMMAs.
__global__ void __launch_bounds__(256)
gemm_kernel(const _Float16* __restrict__ A, const _Float16* __restrict__ Bt,
            const float* __restrict__ bias, float* __restrict__ C) {
  __shared__ __align__(32) _Float16 Btile[2][64 * 64];   // [buf][col*64 + k]
  int tid  = threadIdx.x;
  int lane = tid & 31;
  int w    = tid >> 5;                  // wave 0..7
  int mblk = blockIdx.x >> 4;           // 32 row-blocks of 256
  int ntb  = blockIdx.x & 15;           // 16 col-blocks of 64
  int lm = lane & 15, hi8 = (lane & 16) ? 8 : 0;
  int row0 = mblk * 256 + w * 32;       // this wave's 32-row slab

  // this thread's slice of the 8KB B tile: 32B (2 x b128) of column `col`
  int col  = tid >> 2;                  // 0..63
  int part = tid & 3;                   // 4 x 32B per 128B column
  const _Float16* gsrc0 = Bt + (size_t)(ntb * 64 + col) * D + part * 16;
  unsigned lb0 = lds_addr_of(&Btile[0][col * 64 + part * 16]);
  unsigned lb1 = lds_addr_of(&Btile[1][col * 64 + part * 16]);

  const _Float16* arow0 = A + (size_t)(row0 + lm) * D;
  const _Float16* arow1 = A + (size_t)(row0 + 16 + lm) * D;
  v8f a00 = {}, a01 = {}, a02 = {}, a03 = {};
  v8f a10 = {}, a11 = {}, a12 = {}, a13 = {};

  // prologue: DMA tile 0
  async_load_lds_b128(lb0,      gsrc0);
  async_load_lds_b128(lb0 + 16, gsrc0 + 8);

  for (int it = 0; it < 16; ++it) {
    wait_async0();          // my DMA done; barrier => everyone's DMA done and
    __syncthreads();        //   everyone finished reading the other buffer
    if (it + 1 < 16) {      // kick DMA for the next 64-K tile, other buffer
      const _Float16* g = gsrc0 + (size_t)(it + 1) * 64;
      unsigned l = (it & 1) ? lb0 : lb1;
      async_load_lds_b128(l,      g);
      async_load_lds_b128(l + 16, g + 8);
    }
    const _Float16* bl = &Btile[it & 1][0];
    int k0 = it * 64;
    __builtin_prefetch(arow0 + k0 + 256, 0, 0);  // -> global_prefetch_b8
#pragma unroll
    for (int su = 0; su < 2; ++su) {             // two K=32 sub-steps
      int kk = k0 + su * 32;
      v16h aA0 = load_afrag(arow0 + kk, lane);
      v16h aA1 = load_afrag(arow1 + kk, lane);
      const _Float16* bs = bl + su * 32;
      v16h b0 = load_bfrag(bs + (0 * 16 + lm) * 64, lane);
      v16h b1 = load_bfrag(bs + (1 * 16 + lm) * 64, lane);
      v16h b2 = load_bfrag(bs + (2 * 16 + lm) * 64, lane);
      v16h b3 = load_bfrag(bs + (3 * 16 + lm) * 64, lane);
      a00 = wmma_f16(aA0, b0, a00);  a10 = wmma_f16(aA1, b0, a10);
      a01 = wmma_f16(aA0, b1, a01);  a11 = wmma_f16(aA1, b1, a11);
      a02 = wmma_f16(aA0, b2, a02);  a12 = wmma_f16(aA1, b2, a12);
      a03 = wmma_f16(aA0, b3, a03);  a13 = wmma_f16(aA1, b3, a13);
    }
  }
#pragma unroll
  for (int r = 0; r < 8; ++r) {
    int m = r + hi8;
    float bv0 = bias[ntb * 64 + 0 * 16 + lm];
    float bv1 = bias[ntb * 64 + 1 * 16 + lm];
    float bv2 = bias[ntb * 64 + 2 * 16 + lm];
    float bv3 = bias[ntb * 64 + 3 * 16 + lm];
    float* d0 = C + (size_t)(row0 + m) * D + ntb * 64;
    d0[0 * 16 + lm] = a00[r] + bv0;
    d0[1 * 16 + lm] = a01[r] + bv1;
    d0[2 * 16 + lm] = a02[r] + bv2;
    d0[3 * 16 + lm] = a03[r] + bv3;
    float* d1 = C + (size_t)(row0 + 16 + m) * D + ntb * 64;
    d1[0 * 16 + lm] = a10[r] + bv0;
    d1[1 * 16 + lm] = a11[r] + bv1;
    d1[2 * 16 + lm] = a12[r] + bv2;
    d1[3 * 16 + lm] = a13[r] + bv3;
  }
}

// ---------------------------------------------------------------------------
// 5) out = LN(x (+ add)) * g + b, optional LeakyReLU, optional f16 mirror.
//    One 256-thread block per 1024-wide row.
__global__ void __launch_bounds__(256)
row_ln_kernel(const float* __restrict__ x, const float* __restrict__ add,
              const float* __restrict__ g, const float* __restrict__ bet,
              float* __restrict__ outf, _Float16* __restrict__ outh,
              int leaky) {
  __shared__ float wsum[16];
  __shared__ float bc[2];
  int tid = threadIdx.x;
  size_t row = blockIdx.x;
  v4f v = *(const v4f*)(x + row * D + tid * 4);
  if (add) {
    v4f a = *(const v4f*)(add + row * D + tid * 4);
#pragma unroll
    for (int i = 0; i < 4; ++i) v[i] += a[i];
  }
  float s = v[0] + v[1] + v[2] + v[3];
  float qq = v[0] * v[0] + v[1] * v[1] + v[2] * v[2] + v[3] * v[3];
  s = redsum32(s);
  qq = redsum32(qq);
  int w = tid >> 5, lane = tid & 31;
  if (lane == 0) { wsum[w] = s; wsum[8 + w] = qq; }
  __syncthreads();
  if (tid == 0) {
    float ts = 0.f, tq = 0.f;
#pragma unroll
    for (int i = 0; i < 8; ++i) { ts += wsum[i]; tq += wsum[8 + i]; }
    float mean = ts * (1.0f / D);
    float var  = tq * (1.0f / D) - mean * mean;
    bc[0] = mean;
    bc[1] = rsqrtf(var + LN_EPS);
  }
  __syncthreads();
  float mean = bc[0], inv = bc[1];
#pragma unroll
  for (int i = 0; i < 4; ++i) {
    int d = tid * 4 + i;
    float y = (v[i] - mean) * inv * g[d] + bet[d];
    if (leaky) y = (y >= 0.0f) ? y : SLOPE * y;
    if (outf) outf[row * D + d] = y;
    if (outh) outh[row * D + d] = (_Float16)y;
  }
}

// ---------------------------------------------------------------------------
extern "C" void kernel_launch(void* const* d_in, const int* in_sizes, int n_in,
                              void* d_out, int out_size, void* d_ws,
                              size_t ws_size, hipStream_t stream) {
  (void)in_sizes; (void)n_in; (void)out_size; (void)ws_size;
  const float* q     = (const float*)d_in[0];
  const int*   masks = (const int*)d_in[1];
  const float* w1    = (const float*)d_in[2];
  const float* b1    = (const float*)d_in[3];
  const float* ln1_g = (const float*)d_in[4];
  const float* ln1_b = (const float*)d_in[5];
  const float* w2    = (const float*)d_in[6];
  const float* b2    = (const float*)d_in[7];
  const float* ln2_g = (const float*)d_in[8];
  const float* ln2_b = (const float*)d_in[9];
  const float* anf_g = (const float*)d_in[10];
  const float* anf_b = (const float*)d_in[11];
  const float* an1_g = (const float*)d_in[12];
  const float* an1_b = (const float*)d_in[13];
  const float* an2_g = (const float*)d_in[14];
  const float* an2_b = (const float*)d_in[15];

  constexpr size_t NEL = (size_t)B * S * D;     // 8,388,608
  char* ws = (char*)d_ws;
  size_t off = 0;
  auto take = [&](size_t bytes) -> void* {
    void* p = ws + off;
    off += (bytes + 255) & ~(size_t)255;
    return p;
  };
  _Float16* qn   = (_Float16*)take(NEL * 2);            // (B,H,S,DH) f16
  _Float16* vt   = (_Float16*)take(NEL * 2);            // (B,H,DH,S) f16
  float*    attn = (float*)   take(NEL * 4);
  float*    y1f  = (float*)   take(NEL * 4);
  _Float16* y1h  = (_Float16*)take(NEL * 2);
  _Float16* w1t  = (_Float16*)take((size_t)D * D * 2);
  _Float16* w2t  = (_Float16*)take((size_t)D * D * 2);
  float*    g1   = (float*)   take(NEL * 4);
  _Float16* hh   = (_Float16*)take(NEL * 2);
  float*    g2   = (float*)   take(NEL * 4);
  float*    yv   = (float*)   take(NEL * 4);
  float*    y2f  = (float*)   take(NEL * 4);

  const int ROWS = B * S;                       // 8192 LN rows

  prep_kernel<<<(B * H * S) / 8, 256, 0, stream>>>(q, qn, vt);
  conv_wt_kernel<<<(D * D) / 256, 256, 0, stream>>>(w1, w1t);
  conv_wt_kernel<<<(D * D) / 256, 256, 0, stream>>>(w2, w2t);
  // one wave per (b, h, 16-query block)
  attn_kernel<<<B * H * (S / 16), 32, 0, stream>>>(qn, vt, masks, attn);
  // y1 = LN(q + attn)
  row_ln_kernel<<<ROWS, 256, 0, stream>>>(q, attn, an1_g, an1_b, y1f, y1h, 0);
  // g1 = y1 @ w1 + b1 ; h = leaky(LN(g1))
  gemm_kernel<<<(ROWS / 256) * (D / 64), 256, 0, stream>>>(y1h, w1t, b1, g1);
  row_ln_kernel<<<ROWS, 256, 0, stream>>>(g1, nullptr, ln1_g, ln1_b, nullptr, hh, 1);
  // g2 = h @ w2 + b2 ; y = leaky(LN(g2))
  gemm_kernel<<<(ROWS / 256) * (D / 64), 256, 0, stream>>>(hh, w2t, b2, g2);
  row_ln_kernel<<<ROWS, 256, 0, stream>>>(g2, nullptr, ln2_g, ln2_b, yv, nullptr, 1);
  // y2 = LN(y1 + y) ; out = LN(y1 + y2)
  row_ln_kernel<<<ROWS, 256, 0, stream>>>(y1f, yv, anf_g, anf_b, y2f, nullptr, 0);
  row_ln_kernel<<<ROWS, 256, 0, stream>>>(y1f, y2f, an2_g, an2_b, (float*)d_out,
                                          nullptr, 0);
}
